// PoseFeatureNet_23819888624117
// MI455X (gfx1250) — matrix operand order repaired and checked by reading, not compiled, not run on hardware
//
#include <hip/hip_runtime.h>
#include <hip/hip_bf16.h>
#include <math.h>

#define BB 128
#define TT 12
#define JJ 17
#define NJ (BB*JJ)        // 2176 nodes per timestep
#define HID 128
#define HEADS 8
#define HH (HEADS*HID)    // 1024
#define EIN 2432
#define LHS 512
#define G4 (4*LHS)        // 2048
#define NCLS 500
#define NCLSP 512         // padded classifier N
#define NE 38
#define ROWS 256          // 2B
#define MROW (ROWS*TT)    // 3072
#define LPITCH 40         // LDS pitch (halfwords)
#define LBUF (128*LPITCH) // one LDS tile buffer

// CONN: row0 = sources, row1 = targets (19 pairs + reversed)
__constant__ int d_c0[NE] = {15,13,16,14,11,5,6,5,5,6,7,8,1,0,0,1,2,3,4,
                             13,11,14,12,12,11,12,6,7,8,9,10,2,1,2,3,4,5,6};
__constant__ int d_c1[NE] = {13,11,14,12,12,11,12,6,7,8,9,10,2,1,2,3,4,5,6,
                             15,13,16,14,11,5,6,5,5,6,7,8,1,0,0,1,2,3,4};

typedef __attribute__((ext_vector_type(16))) __bf16 bf16x16;
typedef __attribute__((ext_vector_type(8)))  float  f32x8;

union FragU { bf16x16 v; unsigned int u[8]; };

__device__ __forceinline__ unsigned short f2bf(float x) {
  union { float f; unsigned int u; } v; v.f = x;
  unsigned int r = v.u + 0x7FFFu + ((v.u >> 16) & 1u);   // RNE
  return (unsigned short)(r >> 16);
}
__device__ __forceinline__ float sigf(float x) { return 1.f / (1.f + expf(-x)); }
__device__ __forceinline__ float eluf(float x) { return x > 0.f ? x : (expf(x) - 1.f); }
__device__ __forceinline__ float lrelu(float x) { return x > 0.f ? x : 0.2f * x; }

// ---------------------------------------------------------------------------
// bf16 WMMA GEMM: C[M,N] f32 (+optional per-col bias) = A[M,K] * B^T
//   B given as [Npad,K] row-major (weights pre-transposed / padded).
// Requirements: M % 128 == 0, K % 32 == 0, staged Npad covers the grid.
// N is the true col count (epilogue bound only).
// Block: 256 threads = 8 waves; tile 128x128x32; wave tile 32x64 = 8 WMMAs.
// LDS double-buffered: fetch tile k+1 into regs while WMMAs consume tile k,
// single barrier per K-step.
// ---------------------------------------------------------------------------
__global__ void __launch_bounds__(256)
wmma_gemm_kernel(const unsigned short* __restrict__ A,
                 const unsigned short* __restrict__ Bm,
                 float* __restrict__ C,
                 const float* __restrict__ bias,
                 int M, int N, int K, int ldc) {
  __shared__ unsigned short As[2 * LBUF];
  __shared__ unsigned short Bs[2 * LBUF];

  const int tid  = threadIdx.x;
  const int lane = tid & 31;
  const int wave = tid >> 5;
  const int wm   = wave & 3;      // 4 row strips of 32
  const int wn   = wave >> 2;     // 2 col strips of 64
  const int m0   = blockIdx.y * 128;
  const int n0   = blockIdx.x * 128;
  const int half = lane >> 4;
  const int l16  = lane & 15;

  f32x8 acc[2][4];
#pragma unroll
  for (int am = 0; am < 2; ++am)
#pragma unroll
    for (int bn = 0; bn < 4; ++bn)
#pragma unroll
      for (int i = 0; i < 8; ++i) acc[am][bn][i] = 0.f;

  const int sr  = tid >> 1;         // 0..127 staged row
  const int skq = (tid & 1) * 16;   // k chunk 0/16
  const int soff = sr * LPITCH + skq;

  const unsigned short* Arow = A  + (size_t)(m0 + sr) * K + skq;
  const unsigned short* Brow = Bm + (size_t)(n0 + sr) * K + skq;

  const int steps = K >> 5;

  // Preload tile 0 into buffer 0
  {
    const uint4* sa = (const uint4*)(Arow);
    const uint4* sb = (const uint4*)(Brow);
    uint4 a0 = sa[0], a1 = sa[1];
    uint4 b0 = sb[0], b1 = sb[1];
    uint4* ad = (uint4*)(As + soff);
    uint4* bd = (uint4*)(Bs + soff);
    ad[0] = a0; ad[1] = a1;
    bd[0] = b0; bd[1] = b1;
  }
  __syncthreads();

  for (int kk = 0; kk < steps; ++kk) {
    const bool more = (kk + 1) < steps;
    uint4 na0 = make_uint4(0u,0u,0u,0u), na1 = na0, nb0 = na0, nb1 = na0;
    if (more) {
      const unsigned short* ap = Arow + (kk + 1) * 32;
      const unsigned short* bp = Brow + (kk + 1) * 32;
      if (kk + 2 < steps) {               // global_prefetch_b8 the tile after
        __builtin_prefetch(ap + 32, 0, 1);
        __builtin_prefetch(bp + 32, 0, 1);
      }
      const uint4* sa = (const uint4*)ap;
      const uint4* sb = (const uint4*)bp;
      na0 = sa[0]; na1 = sa[1];
      nb0 = sb[0]; nb1 = sb[1];
    }

    // Fragments per ISA layout (CDNA5 ISA, 16-bit A 16x32 / B 32x16)
    const unsigned short* Ab = As + (kk & 1) * LBUF;
    const unsigned short* Bb = Bs + (kk & 1) * LBUF;
    FragU a[2], b[4];
#pragma unroll
    for (int am = 0; am < 2; ++am) {
      const unsigned int* Ar =
          (const unsigned int*)(Ab + (wm * 32 + am * 16 + l16) * LPITCH);
#pragma unroll
      for (int i = 0; i < 4; ++i) {
        a[am].u[i]     = Ar[half * 4 + i];       // K 0..7  / 8..15
        a[am].u[4 + i] = Ar[8 + half * 4 + i];   // K 16..23 / 24..31
      }
    }
#pragma unroll
    for (int bn = 0; bn < 4; ++bn) {
      const unsigned int* Br =
          (const unsigned int*)(Bb + (wn * 64 + bn * 16 + l16) * LPITCH);
#pragma unroll
      for (int i = 0; i < 4; ++i) {
        b[bn].u[i]     = Br[half * 4 + i];       // K 0..7  / 8..15
        b[bn].u[4 + i] = Br[8 + half * 4 + i];   // K 16..23 / 24..31
      }
    }
#pragma unroll
    for (int am = 0; am < 2; ++am)
#pragma unroll
      for (int bn = 0; bn < 4; ++bn)
        acc[am][bn] = __builtin_amdgcn_wmma_f32_16x16x32_bf16(
            false, a[am].v, false, b[bn].v, (short)0, acc[am][bn], false, false);

    if (more) {
      uint4* ad = (uint4*)(As + ((kk + 1) & 1) * LBUF + soff);
      uint4* bd = (uint4*)(Bs + ((kk + 1) & 1) * LBUF + soff);
      ad[0] = na0; ad[1] = na1;
      bd[0] = nb0; bd[1] = nb1;
    }
    __syncthreads();
  }

  // Epilogue: C lane layout M = i + 8*half, N = l16
#pragma unroll
  for (int am = 0; am < 2; ++am) {
    int rb = m0 + wm * 32 + am * 16 + half * 8;
#pragma unroll
    for (int bn = 0; bn < 4; ++bn) {
      int cc = n0 + wn * 64 + bn * 16 + l16;
      if (cc < N) {
        float bia = bias ? bias[cc] : 0.f;
#pragma unroll
        for (int i = 0; i < 8; ++i)
          C[(size_t)(rb + i) * ldc + cc] = acc[am][bn][i] + bia;
      }
    }
  }
}

// ---------------------------------------------------------------------------
// GAT layer 1: h = x @ W1 (K=3); self-loop-only nodes finalize elu(h+b1);
// store h rows + attention logits for the 17 edge-connected nodes.
// ---------------------------------------------------------------------------
__global__ void __launch_bounds__(256)
gat1_kernel(const float* __restrict__ pose, const float* __restrict__ W1,
            const float* __restrict__ a_src, const float* __restrict__ a_dst,
            const float* __restrict__ b1,
            unsigned short* __restrict__ hcat, float* __restrict__ h17,
            float* __restrict__ als, float* __restrict__ ald, int t) {
  int n = blockIdx.x, tid = threadIdx.x;
  int b = n / JJ, j = n % JJ;
  const float* xp = pose + ((size_t)(b * TT + t) * JJ + j) * 3;
  float x0 = xp[0], x1 = xp[1], x2 = xp[2];
  __shared__ float s_als[HEADS], s_ald[HEADS];
  if (tid < HEADS) { s_als[tid] = 0.f; s_ald[tid] = 0.f; }
  __syncthreads();
#pragma unroll
  for (int k = 0; k < 4; ++k) {
    int c = tid + k * 256;
    float h = x0 * W1[c] + x1 * W1[HH + c] + x2 * W1[2 * HH + c];
    if (n < JJ) {
      h17[n * HH + c] = h;
      int hh = c >> 7, d = c & 127;
      atomicAdd(&s_als[hh], h * a_src[hh * HID + d]);
      atomicAdd(&s_ald[hh], h * a_dst[hh * HID + d]);
    }
    hcat[(size_t)n * HH + c] = f2bf(eluf(h + b1[c]));
  }
  __syncthreads();
  if (n < JJ && tid < HEADS) {
    als[n * HEADS + tid] = s_als[tid];
    ald[n * HEADS + tid] = s_ald[tid];
  }
}

// Attention fix-up for targets 0..16 of layer 1 (one block per target)
__global__ void __launch_bounds__(256)
gat_fix1_kernel(const float* __restrict__ h17, const float* __restrict__ als,
                const float* __restrict__ ald, const float* __restrict__ b1,
                unsigned short* __restrict__ hcat) {
  int n = blockIdx.x, tid = threadIdx.x;
  __shared__ float w_e[NE * HEADS], w_self[HEADS];
  if (tid < HEADS) {
    int h = tid;
    float eself = lrelu(als[n * HEADS + h] + ald[n * HEADS + h]);
    float m = eself;
    float ev[NE];
    for (int e = 0; e < NE; ++e) {
      if (d_c1[e] == n) {
        float x = lrelu(als[d_c0[e] * HEADS + h] + ald[n * HEADS + h]);
        ev[e] = x; if (x > m) m = x;
      } else ev[e] = -1e30f;
    }
    float den = expf(eself - m);
    float wself = den;
    for (int e = 0; e < NE; ++e) {
      float xx = (ev[e] > -1e29f) ? expf(ev[e] - m) : 0.f;
      w_e[e * HEADS + h] = xx; den += xx;
    }
    float inv = 1.f / den;
    w_self[h] = wself * inv;
    for (int e = 0; e < NE; ++e) w_e[e * HEADS + h] *= inv;
  }
  __syncthreads();
#pragma unroll
  for (int k = 0; k < 4; ++k) {
    int c = tid + k * 256;
    int h = c >> 7;
    float acc = w_self[h] * h17[n * HH + c];
    for (int e = 0; e < NE; ++e) acc += w_e[e * HEADS + h] * h17[d_c0[e] * HH + c];
    hcat[(size_t)n * HH + c] = f2bf(eluf(acc + b1[c]));
  }
}

// Layer-2 attention logits for nodes 0..16
__global__ void __launch_bounds__(256)
gat2_al_kernel(const float* __restrict__ H2, const float* __restrict__ a_src,
               const float* __restrict__ a_dst, float* __restrict__ als,
               float* __restrict__ ald) {
  int n = blockIdx.x, tid = threadIdx.x;
  __shared__ float s_als[HEADS], s_ald[HEADS];
  if (tid < HEADS) { s_als[tid] = 0.f; s_ald[tid] = 0.f; }
  __syncthreads();
#pragma unroll
  for (int k = 0; k < 4; ++k) {
    int c = tid + k * 256;
    float h = H2[(size_t)n * HH + c];
    int hh = c >> 7, d = c & 127;
    atomicAdd(&s_als[hh], h * a_src[hh * HID + d]);
    atomicAdd(&s_ald[hh], h * a_dst[hh * HID + d]);
  }
  __syncthreads();
  if (tid < HEADS) { als[n * HEADS + tid] = s_als[tid]; ald[n * HEADS + tid] = s_ald[tid]; }
}

// Layer-2 aggregation + head-mean + b2 -> feats (one block per node, 128 thr)
__global__ void __launch_bounds__(128)
gat2_post_kernel(const float* __restrict__ H2, const float* __restrict__ als,
                 const float* __restrict__ ald, const float* __restrict__ b2,
                 float* __restrict__ feats, int p, int t) {
  int n = blockIdx.x, tid = threadIdx.x;
  __shared__ float w_e[NE * HEADS], w_self[HEADS];
  if (n < JJ && tid < HEADS) {
    int h = tid;
    float eself = lrelu(als[n * HEADS + h] + ald[n * HEADS + h]);
    float m = eself;
    float ev[NE];
    for (int e = 0; e < NE; ++e) {
      if (d_c1[e] == n) {
        float x = lrelu(als[d_c0[e] * HEADS + h] + ald[n * HEADS + h]);
        ev[e] = x; if (x > m) m = x;
      } else ev[e] = -1e30f;
    }
    float den = expf(eself - m);
    float wself = den;
    for (int e = 0; e < NE; ++e) {
      float xx = (ev[e] > -1e29f) ? expf(ev[e] - m) : 0.f;
      w_e[e * HEADS + h] = xx; den += xx;
    }
    float inv = 1.f / den;
    w_self[h] = wself * inv;
    for (int e = 0; e < NE; ++e) w_e[e * HEADS + h] *= inv;
  }
  __syncthreads();
  int b = n / JJ, j = n % JJ;
  float acc = 0.f;
  if (n >= JJ) {
    for (int h = 0; h < HEADS; ++h) acc += H2[(size_t)n * HH + h * HID + tid];
  } else {
    for (int h = 0; h < HEADS; ++h) {
      float s = w_self[h] * H2[(size_t)n * HH + h * HID + tid];
      for (int e = 0; e < NE; ++e)
        s += w_e[e * HEADS + h] * H2[(size_t)d_c0[e] * HH + h * HID + tid];
      acc += s;
    }
  }
  acc = acc * (1.f / HEADS) + b2[tid];
  feats[((size_t)(p * BB + b) * TT + t) * EIN + j * HID + tid] = acc;
}

// Edge branch: lengths/angles of 38 bones -> 2x128 features per (b,t)
__global__ void __launch_bounds__(256)
edge_kernel(const float* __restrict__ pose, const float* __restrict__ W_fc,
            const float* __restrict__ b_fc, float* __restrict__ feats, int p) {
  int bt = blockIdx.x;
  int b = bt / TT, t = bt % TT;
  int tid = threadIdx.x;
  __shared__ float ef[2 * NE];
  if (tid < 2 * NE) {
    int e = tid >> 1, comp = tid & 1;
    const float* pb = pose + ((size_t)(b * TT + t) * JJ) * 3;
    float vx = pb[d_c1[e] * 3 + 0] - pb[d_c0[e] * 3 + 0];
    float vy = pb[d_c1[e] * 3 + 1] - pb[d_c0[e] * 3 + 1];
    ef[tid] = comp ? atan2f(vy, vx) : sqrtf(vx * vx + vy * vy);
  }
  __syncthreads();
  int r = tid >> 7, c = tid & 127;
  float acc = b_fc[c];
  for (int k = 0; k < NE; ++k) acc += ef[r * NE + k] * W_fc[k * HID + c];
  feats[((size_t)(p * BB + b) * TT + t) * EIN + JJ * HID + r * HID + c] = acc;
}

// BatchNorm over T channel + bf16 cast for LSTM GEMM
__global__ void bn_kernel(const float* __restrict__ feats, const float* __restrict__ g,
                          const float* __restrict__ be, const float* __restrict__ mu,
                          const float* __restrict__ var, unsigned short* __restrict__ out,
                          int total) {
  int idx = blockIdx.x * 256 + threadIdx.x;
  if (idx >= total) return;
  int t = (idx / EIN) % TT;
  float y = (feats[idx] - mu[t]) * rsqrtf(var[t] + 1e-5f) * g[t] + be[t];
  out[idx] = f2bf(y);
}

__global__ void lstm_init_kernel(unsigned short* hbf, float* cbuf) {
  int idx = blockIdx.x * 256 + threadIdx.x;
  if (idx < ROWS * LHS) { hbf[idx] = 0; cbuf[idx] = 0.f; }
}

__global__ void lstm_gate_kernel(const float* __restrict__ pre,
                                 const float* __restrict__ gtmp,
                                 float* __restrict__ cbuf,
                                 unsigned short* __restrict__ hbf,
                                 float* __restrict__ lout, int dir, int tt) {
  int idx = blockIdx.x * 256 + threadIdx.x;
  if (idx >= ROWS * LHS) return;
  int r = idx >> 9, u = idx & (LHS - 1);
  const float* pr = pre + (size_t)(r * TT + tt) * G4;
  const float* gt = gtmp + (size_t)r * G4;
  float gi = pr[u] + gt[u];
  float gf = pr[LHS + u] + gt[LHS + u];
  float gg = pr[2 * LHS + u] + gt[2 * LHS + u];
  float go = pr[3 * LHS + u] + gt[3 * LHS + u];
  float c = sigf(gf) * cbuf[idx] + sigf(gi) * tanhf(gg);
  float h = sigf(go) * tanhf(c);
  cbuf[idx] = c;
  hbf[idx] = f2bf(h);
  lout[(size_t)(r * TT + tt) * (2 * LHS) + dir * LHS + u] = h;
}

// Temporal attention pooling: one block per batch row
__global__ void __launch_bounds__(256)
attn_kernel(const float* __restrict__ lout, const float* __restrict__ W_att,
            const float* __restrict__ b_att, float* __restrict__ out_att,
            unsigned short* __restrict__ att_bf) {
  int r = blockIdx.x, tid = threadIdx.x;
  __shared__ float red[256];
  __shared__ float sc[TT];
  __shared__ float wts[TT];
  for (int t = 0; t < TT; ++t) {
    float part = 0.f;
#pragma unroll
    for (int k = 0; k < 4; ++k) {
      int d = tid + k * 256;
      part += lout[(size_t)(r * TT + t) * 1024 + d] * W_att[d];
    }
    red[tid] = part; __syncthreads();
    for (int s = 128; s > 0; s >>= 1) {
      if (tid < s) red[tid] += red[tid + s];
      __syncthreads();
    }
    if (tid == 0) sc[t] = red[0] + b_att[0];
    __syncthreads();
  }
  if (tid == 0) {
    float m = sc[0];
    for (int t = 1; t < TT; ++t) m = fmaxf(m, sc[t]);
    float den = 0.f;
    for (int t = 0; t < TT; ++t) { wts[t] = expf(sc[t] - m); den += wts[t]; }
    float inv = 1.f / den;
    for (int t = 0; t < TT; ++t) wts[t] *= inv;
  }
  __syncthreads();
#pragma unroll
  for (int k = 0; k < 4; ++k) {
    int d = tid + k * 256;
    float a = 0.f;
    for (int t = 0; t < TT; ++t) a += wts[t] * lout[(size_t)(r * TT + t) * 1024 + d];
    out_att[(size_t)r * 1024 + d] = a;
    att_bf[(size_t)r * 1024 + d] = f2bf(a);
  }
}

__global__ void cvt_kernel(const float* __restrict__ s, unsigned short* __restrict__ d, int n) {
  int i = blockIdx.x * 256 + threadIdx.x;
  if (i < n) d[i] = f2bf(s[i]);
}
// Transpose convert: src [R,C] f32 -> dst [C,R] bf16
__global__ void cvt_t_kernel(const float* __restrict__ s, unsigned short* __restrict__ d,
                             int R, int Cc) {
  int idx = blockIdx.x * 256 + threadIdx.x;
  if (idx >= R * Cc) return;
  int r = idx / Cc, c = idx % Cc;
  d[(size_t)c * R + r] = f2bf(s[idx]);
}
// Convert with row padding: src [Nr,K] f32 -> dst [Npad,K] bf16, zero pad rows
__global__ void cvt_pad_kernel(const float* __restrict__ s, unsigned short* __restrict__ d,
                               int Nr, int Npad, int K) {
  int idx = blockIdx.x * 256 + threadIdx.x;
  if (idx >= Npad * K) return;
  int n = idx / K;
  d[idx] = (n < Nr) ? f2bf(s[idx]) : (unsigned short)0;
}
__global__ void addb_kernel(const float* a, const float* b, float* o, int n) {
  int i = blockIdx.x * 256 + threadIdx.x;
  if (i < n) o[i] = a[i] + b[i];
}

// ---------------------------------------------------------------------------
extern "C" void kernel_launch(void* const* d_in, const int* in_sizes, int n_in,
                              void* d_out, int out_size, void* d_ws, size_t ws_size,
                              hipStream_t stream) {
  const float* pose[2] = { (const float*)d_in[0], (const float*)d_in[1] };
  const float* W1     = (const float*)d_in[2];
  const float* a_src1 = (const float*)d_in[3];
  const float* a_dst1 = (const float*)d_in[4];
  const float* b1     = (const float*)d_in[5];
  const float* W2     = (const float*)d_in[6];
  const float* a_src2 = (const float*)d_in[7];
  const float* a_dst2 = (const float*)d_in[8];
  const float* b2     = (const float*)d_in[9];
  const float* W_fc   = (const float*)d_in[10];
  const float* b_fc   = (const float*)d_in[11];
  const float* bn_g   = (const float*)d_in[12];
  const float* bn_b   = (const float*)d_in[13];
  const float* bn_m   = (const float*)d_in[14];
  const float* bn_v   = (const float*)d_in[15];
  const float* Wih[2] = { (const float*)d_in[16], (const float*)d_in[20] };
  const float* Whh[2] = { (const float*)d_in[17], (const float*)d_in[21] };
  const float* bih[2] = { (const float*)d_in[18], (const float*)d_in[22] };
  const float* bhh[2] = { (const float*)d_in[19], (const float*)d_in[23] };
  const float* W_att  = (const float*)d_in[24];
  const float* b_att  = (const float*)d_in[25];
  const float* W_cls  = (const float*)d_in[26];
  const float* b_cls  = (const float*)d_in[27];

  float* out_att = (float*)d_out;                 // [256,1024]
  float* out_log = (float*)d_out + ROWS * 1024;   // [256,500]

  // Workspace carve-up
  char* base = (char*)d_ws;
  size_t off = 0;
  auto alloc = [&](size_t bytes) -> char* {
    char* p = base + off;
    off = (off + bytes + 255) & ~(size_t)255;
    return p;
  };
  unsigned short* wW2t  = (unsigned short*)alloc((size_t)HH * HH * 2);     // [N,K]
  unsigned short* wWih0 = (unsigned short*)alloc((size_t)G4 * EIN * 2);    // [N,K]
  unsigned short* wWih1 = (unsigned short*)alloc((size_t)G4 * EIN * 2);
  unsigned short* wWhh0 = (unsigned short*)alloc((size_t)G4 * LHS * 2);
  unsigned short* wWhh1 = (unsigned short*)alloc((size_t)G4 * LHS * 2);
  unsigned short* wWcls = (unsigned short*)alloc((size_t)NCLSP * HH * 2);  // padded
  float* biasc0 = (float*)alloc(G4 * 4);
  float* biasc1 = (float*)alloc(G4 * 4);
  unsigned short* hcat = (unsigned short*)alloc((size_t)NJ * HH * 2);
  float* h17  = (float*)alloc((size_t)JJ * HH * 4);
  float* als1 = (float*)alloc(JJ * HEADS * 4);
  float* ald1 = (float*)alloc(JJ * HEADS * 4);
  float* als2 = (float*)alloc(JJ * HEADS * 4);
  float* ald2 = (float*)alloc(JJ * HEADS * 4);
  float* H2   = (float*)alloc((size_t)NJ * HH * 4);
  float* feats = (float*)alloc((size_t)ROWS * TT * EIN * 4);
  unsigned short* feats_bf = (unsigned short*)alloc((size_t)MROW * EIN * 2);
  float* pre0 = (float*)alloc((size_t)MROW * G4 * 4);
  float* pre1 = (float*)alloc((size_t)MROW * G4 * 4);
  float* gtmp = (float*)alloc((size_t)ROWS * G4 * 4);
  unsigned short* hbf = (unsigned short*)alloc((size_t)ROWS * LHS * 2);
  float* cbuf = (float*)alloc((size_t)ROWS * LHS * 4);
  float* lout = (float*)alloc((size_t)ROWS * TT * 1024 * 4);
  unsigned short* att_bf = (unsigned short*)alloc((size_t)ROWS * 1024 * 2);
  unsigned short* wWih[2] = { wWih0, wWih1 };
  unsigned short* wWhh[2] = { wWhh0, wWhh1 };
  float* pre[2]    = { pre0, pre1 };
  float* biasc[2]  = { biasc0, biasc1 };

  auto cvt = [&](const float* s, unsigned short* d, int n) {
    cvt_kernel<<<(n + 255) / 256, 256, 0, stream>>>(s, d, n);
  };
  // W2 [K,N] -> bf16 [N,K] (one-time transpose so GEMM staging is branch-free)
  cvt_t_kernel<<<(HH * HH + 255) / 256, 256, 0, stream>>>(W2, wW2t, HH, HH);
  cvt(Wih[0], wWih0, G4 * EIN);
  cvt(Wih[1], wWih1, G4 * EIN);
  cvt(Whh[0], wWhh0, G4 * LHS);
  cvt(Whh[1], wWhh1, G4 * LHS);
  cvt_pad_kernel<<<(NCLSP * HH + 255) / 256, 256, 0, stream>>>(W_cls, wWcls,
                                                              NCLS, NCLSP, HH);
  addb_kernel<<<(G4 + 255) / 256, 256, 0, stream>>>(bih[0], bhh[0], biasc0, G4);
  addb_kernel<<<(G4 + 255) / 256, 256, 0, stream>>>(bih[1], bhh[1], biasc1, G4);

  // GAT branch, per pose, per timestep
  for (int p = 0; p < 2; ++p) {
    for (int t = 0; t < TT; ++t) {
      gat1_kernel<<<NJ, 256, 0, stream>>>(pose[p], W1, a_src1, a_dst1, b1,
                                          hcat, h17, als1, ald1, t);
      gat_fix1_kernel<<<JJ, 256, 0, stream>>>(h17, als1, ald1, b1, hcat);
      wmma_gemm_kernel<<<dim3(HH / 128, NJ / 128), 256, 0, stream>>>(
          hcat, wW2t, H2, nullptr, NJ, HH, HH, HH);
      gat2_al_kernel<<<JJ, 256, 0, stream>>>(H2, a_src2, a_dst2, als2, ald2);
      gat2_post_kernel<<<NJ, 128, 0, stream>>>(H2, als2, ald2, b2, feats, p, t);
    }
    edge_kernel<<<BB * TT, 256, 0, stream>>>(pose[p], W_fc, b_fc, feats, p);
  }

  // BatchNorm + bf16 cast
  {
    int total = ROWS * TT * EIN;
    bn_kernel<<<(total + 255) / 256, 256, 0, stream>>>(feats, bn_g, bn_b, bn_m, bn_v,
                                                       feats_bf, total);
  }

  // LSTM input projections (X @ Wih^T + bih + bhh)
  for (int d = 0; d < 2; ++d) {
    wmma_gemm_kernel<<<dim3(G4 / 128, MROW / 128), 256, 0, stream>>>(
        feats_bf, wWih[d], pre[d], biasc[d], MROW, G4, EIN, G4);
  }

  // LSTM recurrence
  for (int d = 0; d < 2; ++d) {
    lstm_init_kernel<<<(ROWS * LHS + 255) / 256, 256, 0, stream>>>(hbf, cbuf);
    for (int s = 0; s < TT; ++s) {
      int tt = d ? (TT - 1 - s) : s;
      wmma_gemm_kernel<<<dim3(G4 / 128, ROWS / 128), 256, 0, stream>>>(
          hbf, wWhh[d], gtmp, nullptr, ROWS, G4, LHS, G4);
      lstm_gate_kernel<<<(ROWS * LHS + 255) / 256, 256, 0, stream>>>(
          pre[d], gtmp, cbuf, hbf, lout, d, tt);
    }
  }

  // Attention pooling -> attended (output 0) + bf16 for classifier
  attn_kernel<<<ROWS, 256, 0, stream>>>(lout, W_att, b_att, out_att, att_bf);

  // Classifier: attended @ W_cls^T + b_cls -> logits (output 1)
  wmma_gemm_kernel<<<dim3(NCLSP / 128, ROWS / 128), 256, 0, stream>>>(
      att_bf, wWcls, out_log, b_cls, ROWS, NCLS, HH, NCLS);

  (void)in_sizes; (void)n_in; (void)out_size; (void)ws_size;
}
